// GATPredictor_32744830665084
// MI455X (gfx1250) — matrix operand order repaired and checked by reading, not compiled, not run on hardware
//
#include <hip/hip_runtime.h>
#include <hip/hip_bf16.h>

typedef __attribute__((ext_vector_type(16))) __bf16 v16bf;
typedef __attribute__((ext_vector_type(8)))  float  v8f;

#define HCDIM 256      // H*C
#define NFACT 50000
#define NSTAT 2000
#define DFACT 768
#define NCH   10
#define NEGSL 0.2f

// ---------- helpers ----------
__device__ __forceinline__ unsigned short f2bfu(float f) {
  unsigned u = __float_as_uint(f);
  u += 0x7FFFu + ((u >> 16) & 1u);        // round-to-nearest-even
  return (unsigned short)(u >> 16);
}
__device__ __forceinline__ __bf16 u2bf(unsigned short s) {
  return __builtin_bit_cast(__bf16, s);
}
// order-preserving float -> uint encoding for atomic max
__device__ __forceinline__ unsigned encf(float f) {
  unsigned u = __float_as_uint(f);
  return (u & 0x80000000u) ? ~u : (u | 0x80000000u);
}
__device__ __forceinline__ float decf(unsigned e) {
  return (e & 0x80000000u) ? __uint_as_float(e & 0x7FFFFFFFu)
                           : __uint_as_float(~e);
}

// A tile staged row-major lA[row*32+k]; per-lane elements form two contiguous
// 16B runs -> 2x ds_load_b128.
__device__ __forceinline__ void load_frag_a(const unsigned short* lA, int lane, v16bf& a) {
  int row = lane & 15;
  int kb  = (lane & 16) ? 8 : 0;
#pragma unroll
  for (int j = 0; j < 16; ++j) {
    int vg = j >> 1, w = j & 1;
    int k = (vg < 4) ? (kb + 2 * vg + w) : (16 + kb + 2 * (vg - 4) + w);
    a[j] = u2bf(lA[row * 32 + k]);
  }
}
// B tile staged COLUMN-major lB[col*32+k]; per-lane elements form two
// contiguous 16B runs -> 2x ds_load_b128.
__device__ __forceinline__ void load_frag_b(const unsigned short* lB, int lane, v16bf& b) {
  int col = lane & 15;
  int kb  = (lane & 16) ? 8 : 0;
  const unsigned short* p = lB + col * 32 + kb;
#pragma unroll
  for (int j = 0; j < 16; ++j) b[j] = u2bf(p[(j < 8) ? j : (j + 8)]);
}

// ---------- kernels ----------
__global__ void k_init(float* acc, float* denom, unsigned* mx) {
  int i = blockIdx.x * blockDim.x + threadIdx.x;
  if (i < NFACT * HCDIM) acc[i] = 0.f;
  if (i < NFACT * 2) { denom[i] = 0.f; mx[i] = 0u; }   // 0 == encoded "-huge"
}

// w_ald[k][h] = sum_c Wdst_rm[k, h*128+c] * adst[h,c]
__global__ void k_w_ald(const float* __restrict__ Wdst, const float* __restrict__ adst,
                        float* __restrict__ w_ald) {
  int t = blockIdx.x * blockDim.x + threadIdx.x;
  if (t >= DFACT * 2) return;
  int k = t >> 1, h = t & 1;
  float s = 0.f;
  for (int c = 0; c < 128; ++c) s += Wdst[k * HCDIM + h * 128 + c] * adst[h * 128 + c];
  w_ald[k * 2 + h] = s;
}

// hs = x_statute @ Wsrc_rm  via WMMA bf16, one wave per 16x16 tile
__global__ void k_hs_wmma(const float* __restrict__ xs, const float* __restrict__ W,
                          float* __restrict__ hs) {
  __shared__ unsigned short lA[16 * 32];   // [row][k]
  __shared__ unsigned short lB[16 * 32];   // [col][k]  (column-major tile)
  const int lane = threadIdx.x;
  const int m0 = blockIdx.x * 16, n0 = blockIdx.y * 16;
  v8f acc = {};
  for (int kt = 0; kt < HCDIM / 32; ++kt) {
#pragma unroll
    for (int t = 0; t < 16; ++t) {
      int i = lane + 32 * t;
      { int row = i >> 5, k = i & 31;
        lA[i] = f2bfu(xs[(size_t)(m0 + row) * HCDIM + kt * 32 + k]); }
      { int kk = i >> 4, n = i & 15;    // coalesced global read, scattered LDS write
        lB[n * 32 + kk] = f2bfu(W[(size_t)(kt * 32 + kk) * HCDIM + n0 + n]); }
    }
    __syncthreads();
    v16bf a, b;
    load_frag_a(lA, lane, a);
    load_frag_b(lB, lane, b);
    acc = __builtin_amdgcn_wmma_f32_16x16x32_bf16(false, a, false, b,
                                                  (short)0, acc, false, false);
    __syncthreads();
  }
  const int N = lane & 15, Mb = (lane & 16) ? 8 : 0;
#pragma unroll
  for (int r = 0; r < 8; ++r)
    hs[(size_t)(m0 + Mb + r) * HCDIM + n0 + N] = acc[r];
}

__global__ void k_als(const float* __restrict__ hs, const float* __restrict__ asrc,
                      float* __restrict__ als) {
  int t = blockIdx.x * blockDim.x + threadIdx.x;
  if (t >= NSTAT * 2) return;
  int n = t >> 1, h = t & 1;
  float s = 0.f;
  for (int c = 0; c < 128; ++c) s += hs[(size_t)n * HCDIM + h * 128 + c] * asrc[h * 128 + c];
  als[t] = s;
}

// ald = x_fact @ w_ald  (memory-bound GEMV, one wave per row)
__global__ void k_ald(const float* __restrict__ xf, const float* __restrict__ w_ald,
                      float* __restrict__ ald) {
  __shared__ float w0[DFACT], w1[DFACT];
  for (int i = threadIdx.x; i < DFACT; i += blockDim.x) {
    w0[i] = w_ald[2 * i]; w1[i] = w_ald[2 * i + 1];
  }
  __syncthreads();
  const int lane = threadIdx.x & 31;
  const int n = blockIdx.x * 8 + (threadIdx.x >> 5);
  float a0 = 0.f, a1 = 0.f;
  const float* row = xf + (size_t)n * DFACT;
  for (int k = lane; k < DFACT; k += 32) {
    __builtin_prefetch(row + k + 256, 0, 1);
    float x = row[k];
    a0 += x * w0[k]; a1 += x * w1[k];
  }
#pragma unroll
  for (int off = 16; off > 0; off >>= 1) {
    a0 += __shfl_xor(a0, off, 32);
    a1 += __shfl_xor(a1, off, 32);
  }
  if (lane == 0) { ald[n * 2] = a0; ald[n * 2 + 1] = a1; }
}

__global__ void k_edge1(const int* __restrict__ src, const int* __restrict__ dst,
                        const float* __restrict__ als, const float* __restrict__ ald,
                        float* __restrict__ ebuf, unsigned* __restrict__ mx, int E) {
  int e = blockIdx.x * blockDim.x + threadIdx.x;
  if (e >= E) return;
  int s = src[e], d = dst[e];
#pragma unroll
  for (int h = 0; h < 2; ++h) {
    float v = als[s * 2 + h] + ald[d * 2 + h];
    v = (v > 0.f) ? v : NEGSL * v;
    ebuf[(size_t)e * 2 + h] = v;
    atomicMax(&mx[d * 2 + h], encf(v));
  }
}

__global__ void k_edge2(const int* __restrict__ dst, float* __restrict__ ebuf,
                        const unsigned* __restrict__ mx, float* __restrict__ denom, int E) {
  int e = blockIdx.x * blockDim.x + threadIdx.x;
  if (e >= E) return;
  int d = dst[e];
#pragma unroll
  for (int h = 0; h < 2; ++h) {
    float m = decf(mx[d * 2 + h]);
    float x = __expf(ebuf[(size_t)e * 2 + h] - m);
    ebuf[(size_t)e * 2 + h] = x;
    atomicAdd(&denom[d * 2 + h], x);
  }
}

// one wave per edge; 32 lanes x 8 channels, coalesced atomic scatter into acc (fits in L2)
__global__ void k_edge3(const int* __restrict__ src, const int* __restrict__ dst,
                        const float* __restrict__ ebuf, const float* __restrict__ denom,
                        const float* __restrict__ hs, float* __restrict__ acc, int E) {
  int e = blockIdx.x * 8 + (threadIdx.x >> 5);
  if (e >= E) return;
  const int lane = threadIdx.x & 31;
  int s = src[e], d = dst[e];
  float a0 = ebuf[(size_t)e * 2]     / (denom[d * 2]     + 1e-16f);
  float a1 = ebuf[(size_t)e * 2 + 1] / (denom[d * 2 + 1] + 1e-16f);
  const float* hr = hs + (size_t)s * HCDIM;
  float* ar = acc + (size_t)d * HCDIM;
#pragma unroll
  for (int j = 0; j < 8; ++j) {
    int c = lane + 32 * j;
    float al = (c >= 128) ? a1 : a0;
    atomicAdd(&ar[c], al * hr[c]);
  }
}

// out = (acc + b_rm) @ Wc + bc  via WMMA bf16 (Wc padded to N=16; biases folded)
__global__ void k_out_wmma(const float* __restrict__ acc, const float* __restrict__ Wc,
                           const float* __restrict__ bc, const float* __restrict__ b_rm,
                           float* __restrict__ out) {
  __shared__ unsigned short lB[8][16 * 32];  // per K-tile, [col][k] column-major
  __shared__ unsigned short lA[16 * 32];
  __shared__ float bc2[16];
  const int lane = threadIdx.x;
  for (int i = lane; i < HCDIM * 16; i += 32) {
    int kg = i >> 4, n = i & 15;
    lB[kg >> 5][n * 32 + (kg & 31)] =
        (n < NCH) ? f2bfu(Wc[kg * NCH + n]) : (unsigned short)0;
  }
  if (lane < 16) {
    float s = 0.f;
    if (lane < NCH) {
      s = bc[lane];
      for (int k = 0; k < HCDIM; ++k) s += b_rm[k] * Wc[k * NCH + lane];
    }
    bc2[lane] = s;
  }
  __syncthreads();
  const int m0 = blockIdx.x * 16;
  v8f dacc = {};
  for (int kt = 0; kt < HCDIM / 32; ++kt) {
#pragma unroll
    for (int t = 0; t < 16; ++t) {
      int i = lane + 32 * t;
      int row = i >> 5, k = i & 31;
      lA[i] = f2bfu(acc[(size_t)(m0 + row) * HCDIM + kt * 32 + k]);
    }
    __syncthreads();
    v16bf a, b;
    load_frag_a(lA, lane, a);
    load_frag_b(lB[kt], lane, b);
    dacc = __builtin_amdgcn_wmma_f32_16x16x32_bf16(false, a, false, b,
                                                   (short)0, dacc, false, false);
    __syncthreads();
  }
  const int N = lane & 15, Mb = (lane & 16) ? 8 : 0;
  if (N < NCH) {
#pragma unroll
    for (int r = 0; r < 8; ++r)
      out[(size_t)(m0 + Mb + r) * NCH + N] = dacc[r] + bc2[N];
  }
}

// ---------- launch ----------
extern "C" void kernel_launch(void* const* d_in, const int* in_sizes, int n_in,
                              void* d_out, int out_size, void* d_ws, size_t ws_size,
                              hipStream_t stream) {
  (void)n_in; (void)out_size; (void)ws_size;
  const float* x_fact  = (const float*)d_in[0];
  const float* x_stat  = (const float*)d_in[1];
  const float* Wsrc_rm = (const float*)d_in[8];
  const float* Wdst_rm = (const float*)d_in[9];
  const float* asrc_rm = (const float*)d_in[10];
  const float* adst_rm = (const float*)d_in[11];
  const float* b_rm    = (const float*)d_in[12];
  const float* Wc      = (const float*)d_in[23];
  const float* bc      = (const float*)d_in[24];
  const int*   rm_src  = (const int*)d_in[27];
  const int*   rm_dst  = (const int*)d_in[28];
  const int E = in_sizes[27];

  char* ws = (char*)d_ws;
  size_t o = 0;
  auto take = [&](size_t bytes) -> char* {
    char* p = ws + o;
    o = (o + bytes + 255) & ~(size_t)255;
    return p;
  };
  float*    w_ald = (float*)take((size_t)DFACT * 2 * 4);
  float*    hs    = (float*)take((size_t)NSTAT * HCDIM * 4);
  float*    als   = (float*)take((size_t)NSTAT * 2 * 4);
  float*    ald   = (float*)take((size_t)NFACT * 2 * 4);
  float*    ebuf  = (float*)take((size_t)E * 2 * 4);
  unsigned* mx    = (unsigned*)take((size_t)NFACT * 2 * 4);
  float*    denom = (float*)take((size_t)NFACT * 2 * 4);
  float*    acc   = (float*)take((size_t)NFACT * HCDIM * 4);

  k_init<<<(NFACT * HCDIM + 255) / 256, 256, 0, stream>>>(acc, denom, mx);
  k_w_ald<<<(DFACT * 2 + 255) / 256, 256, 0, stream>>>(Wdst_rm, adst_rm, w_ald);
  k_hs_wmma<<<dim3(NSTAT / 16, HCDIM / 16), 32, 0, stream>>>(x_stat, Wsrc_rm, hs);
  k_als<<<(NSTAT * 2 + 255) / 256, 256, 0, stream>>>(hs, asrc_rm, als);
  k_ald<<<NFACT / 8, 256, 0, stream>>>(x_fact, w_ald, ald);
  k_edge1<<<(E + 255) / 256, 256, 0, stream>>>(rm_src, rm_dst, als, ald, ebuf, mx, E);
  k_edge2<<<(E + 255) / 256, 256, 0, stream>>>(rm_dst, ebuf, mx, denom, E);
  k_edge3<<<(E + 7) / 8, 256, 0, stream>>>(rm_src, rm_dst, ebuf, denom, hs, acc, E);
  k_out_wmma<<<NFACT / 16, 32, 0, stream>>>(acc, Wc, bc, b_rm, (float*)d_out);
}
